// RAWTextureDetector_57715770523652
// MI455X (gfx1250) — matrix operand order repaired and checked by reading, not compiled, not run on hardware
//
#include <hip/hip_runtime.h>
#include <hip/hip_bf16.h>
#include <math.h>

typedef __attribute__((ext_vector_type(16))) _Float16 v16h;
typedef __attribute__((ext_vector_type(8)))  _Float16 v8h;
typedef __attribute__((ext_vector_type(8)))  float    v8f;

static __device__ __forceinline__ int refl(int i, int n) {
  // PyTorch 'reflect' padding (edge not repeated); p < n so one reflection suffices
  if (i < 0) i = -i;
  if (i >= n) i = 2 * n - 2 - i;
  return i;
}

__host__ __device__ constexpr int ilog2c(int v) { int r = 0; while (v > 1) { v >>= 1; ++r; } return r; }

// ---------------------------------------------------------------------------
// Weight pre-pack: OIHW f32 -> zero-padded f16 [rows][KPAD], k = tap*CIN + cin
// ---------------------------------------------------------------------------
__global__ void prep_w(const float* __restrict__ w, _Float16* __restrict__ wpad,
                       int cout, int cin_n, int kpad, int csh, int rows)
{
  int i = blockIdx.x * 256 + threadIdx.x;
  if (i >= rows * kpad) return;
  int m = i / kpad, k = i % kpad;
  int cin = k & (cin_n - 1);
  int tap = k >> csh;
  float v = 0.f;
  if (m < cout && tap < 9) v = w[((size_t)m * cin_n + cin) * 9 + tap];
  wpad[i] = (_Float16)v;
}

// ---------------------------------------------------------------------------
// Implicit-GEMM 3x3 conv via V_WMMA_F32_16X16X32_F16.
// Block = 128 threads (4 waves); each wave owns 16 out-channels x 16 pixels of
// one row; the block stages a zero-padded im2col B tile bpatch[64 px][KPAD] in
// LDS so the hot loop is branch-free: 2x global_load_b128 (A, padded weights)
// + 2x ds_load_b128 (B) + 1x v_wmma per 32-wide k-step.
// ---------------------------------------------------------------------------
template<int CIN, int COUT, typename TIN, bool LRELU, bool OUT16, bool OUT32>
__global__ __launch_bounds__(128)
void conv3x3_wmma(const TIN* __restrict__ in, const _Float16* __restrict__ wpad,
                  const float* __restrict__ bias,
                  _Float16* __restrict__ out16, float* __restrict__ out32)
{
  constexpr int H = 512, W = 512;
  constexpr int MT   = (COUT + 15) / 16;
  constexpr int K    = CIN * 9;
  constexpr int KS   = (K + 31) / 32;
  constexpr int KPAD = KS * 32;
  constexpr int SEGS = W / 64;
  constexpr int CSH  = ilog2c(CIN);

  __shared__ __align__(16) _Float16 bpatch[64 * KPAD];

  int idx = blockIdx.x;
  const int xs = idx % SEGS; idx /= SEGS;
  const int y  = idx % H;    idx /= H;
  const int mt = idx % MT;
  const int b  = idx / MT;
  const int x0 = xs * 64;

  // ---- Stage im2col tile: bpatch[xcol*KPAD + k]; branch-free (clamp+select).
  for (int i = threadIdx.x; i < 64 * KPAD; i += 128) {
    int k    = i % KPAD;
    int xcol = i / KPAD;
    int cin  = k & (CIN - 1);
    int tap  = k >> CSH;
    int tapc = tap > 8 ? 8 : tap;
    int kh = tapc / 3, kw = tapc - 3 * kh;
    int gy = y + kh - 1;
    int gx = x0 + xcol + kw - 1;
    int cy = gy < 0 ? 0 : (gy > H - 1 ? H - 1 : gy);
    int cx = gx < 0 ? 0 : (gx > W - 1 ? W - 1 : gx);
    float lv = (float)in[(((size_t)b * CIN + cin) * H + cy) * W + cx];
    bool ok = (tap < 9) && (gy == cy) && (gx == cx);   // zero conv-padding + k-pad
    bpatch[xcol * KPAD + k] = (_Float16)(ok ? lv : 0.f);
  }
  __syncthreads();

  const int lane   = threadIdx.x & 31;
  const int wave   = threadIdx.x >> 5;       // 0..3
  const int lx     = wave * 16;
  const int mbase  = mt * 16;
  const int nn     = lane & 15;
  const int hiHalf = lane >> 4;              // 0 or 1

  // A-fragment rows: padded weights, rows 0..MT*16-1 all valid (zero-padded)
  const _Float16* arow = wpad + (size_t)(mbase + nn) * KPAD;
  const _Float16* brow = bpatch + (size_t)(lx + nn) * KPAD;

  v8f acc = {};
#pragma unroll
  for (int ks = 0; ks < KS; ++ks) {
    const int kbase = ks * 32;
    // A layout (16-bit 16x32): lanes 0-15 -> K 0..7,16..23 ; lanes 16-31 -> +8
    const int ka = kbase + hiHalf * 8;
    v8h a0 = *(const v8h*)(arow + ka);
    v8h a1 = *(const v8h*)(arow + ka + 16);
    v16h a = __builtin_shufflevector(a0, a1, 0, 1, 2, 3, 4, 5, 6, 7,
                                     8, 9, 10, 11, 12, 13, 14, 15);
    // B layout (16-bit 32x16): lanes 0-15 -> K 0..7,16..23 ; lanes 16-31 -> +8
    const int kb = kbase + hiHalf * 8;
    v8h b0 = *(const v8h*)(brow + kb);
    v8h b1 = *(const v8h*)(brow + kb + 16);
    v16h bv = __builtin_shufflevector(b0, b1, 0, 1, 2, 3, 4, 5, 6, 7,
                                      8, 9, 10, 11, 12, 13, 14, 15);
    acc = __builtin_amdgcn_wmma_f32_16x16x32_f16(
        /*neg_a=*/false, a, /*neg_b=*/false, bv,
        /*c_mod=*/(short)0, acc, /*reuse_a=*/false, /*reuse_b=*/false);
  }

  // ---- Store: C/D f32 layout: lanes 0-15 M=r, lanes 16-31 M=r+8 ----------
  const int px   = x0 + lx + nn;
  const int mofs = hiHalf * 8;
#pragma unroll
  for (int r = 0; r < 8; ++r) {
    int ch = mbase + mofs + r;
    if (ch < COUT) {
      float v = acc[r] + bias[ch];
      if (LRELU) v = v >= 0.f ? v : 0.2f * v;
      size_t oi = (((size_t)b * COUT + ch) * H + y) * W + px;
      if (OUT16) out16[oi] = (_Float16)v;
      if (OUT32) out32[oi] = v;
    }
  }
}

// ---------------------------------------------------------------------------
// Separable box filter (reflect padding), one batch image (8 channels) at a time
// ---------------------------------------------------------------------------
__global__ void box_h(const float* __restrict__ f, float* __restrict__ hm,
                      float* __restrict__ hm2, int wsize)
{
  int id = blockIdx.x * blockDim.x + threadIdx.x;
  if (id >= 8 * 512) return;
  int row = id & 511, c = id >> 9;
  const float* fr  = f   + ((size_t)c * 512 + row) * 512;
  float*       om  = hm  + ((size_t)c * 512 + row) * 512;
  float*       om2 = hm2 + ((size_t)c * 512 + row) * 512;
  int p = wsize >> 1;
  float s = 0.f, s2 = 0.f;
  for (int j = -p; j <= p; ++j) { float v = fr[refl(j, 512)]; s += v; s2 += v * v; }
  om[0] = s; om2[0] = s2;
  for (int col = 1; col < 512; ++col) {
    float va = fr[refl(col + p, 512)];
    float vr = fr[refl(col - p - 1, 512)];
    s  += va - vr;
    s2 += va * va - vr * vr;
    om[col] = s; om2[col] = s2;
  }
}

__global__ void box_v(const float* __restrict__ hm, const float* __restrict__ hm2,
                      float* __restrict__ chansum, int wsize)
{
  int id = blockIdx.x * blockDim.x + threadIdx.x;
  if (id >= 8 * 512) return;
  int col = id & 511, c = id >> 9;
  const float* cm  = hm  + (size_t)c * 512 * 512 + col;
  const float* cm2 = hm2 + (size_t)c * 512 * 512 + col;
  int p = wsize >> 1;
  float invw2 = 1.0f / ((float)wsize * (float)wsize);
  float s = 0.f, s2 = 0.f;
  for (int j = -p; j <= p; ++j) {
    int rj = refl(j, 512);
    s += cm[(size_t)rj * 512]; s2 += cm2[(size_t)rj * 512];
  }
  for (int row = 0; row < 512; ++row) {
    if (row) {
      int ia = refl(row + p, 512), ir = refl(row - p - 1, 512);
      s  += cm[(size_t)ia * 512]  - cm[(size_t)ir * 512];
      s2 += cm2[(size_t)ia * 512] - cm2[(size_t)ir * 512];
    }
    float m  = s * invw2, m2 = s2 * invw2;
    float var = m2 - m * m;
    if (var < 1e-6f) var = 1e-6f;
    // softmax(ones(8)) == 1/8 uniform channel weights
    atomicAdd(&chansum[row * 512 + col], 0.125f * sqrtf(var));
  }
}

__global__ void combine_map(float* __restrict__ fused, const float* __restrict__ chansum,
                            const float* __restrict__ fw, int wi)
{
  int i = blockIdx.x * 256 + threadIdx.x;
  if (i < 512 * 512) fused[i] += fw[wi] * powf(chansum[i], 0.8f);
}

__global__ void fill_f32(float* p, int n, float v)
{
  int i = blockIdx.x * blockDim.x + threadIdx.x;
  if (i < n) p[i] = v;
}

__global__ void bias_relu(float* p, const float* fb, int n)
{
  int i = blockIdx.x * blockDim.x + threadIdx.x;
  if (i < n) { float v = p[i] + fb[0]; p[i] = v > 0.f ? v : 0.f; }
}

// ---------------------------------------------------------------------------
// Quantiles via max-reduce + 2048-bin histogram + interpolated CDF
// ---------------------------------------------------------------------------
__global__ void max_reduce(const float* __restrict__ f, int n, unsigned int* outbits)
{
  __shared__ float sm[256];
  float mv = 0.f;
  for (int i = blockIdx.x * blockDim.x + threadIdx.x; i < n; i += gridDim.x * blockDim.x) {
    float v = f[i]; mv = v > mv ? v : mv;
  }
  sm[threadIdx.x] = mv; __syncthreads();
  for (int s = 128; s > 0; s >>= 1) {
    if (threadIdx.x < s) { float o = sm[threadIdx.x + s]; if (o > sm[threadIdx.x]) sm[threadIdx.x] = o; }
    __syncthreads();
  }
  if (threadIdx.x == 0) atomicMax(outbits, __float_as_uint(sm[0])); // values >= 0
}

__global__ void hist_k(const float* __restrict__ f, int n, unsigned int* __restrict__ hist,
                       const unsigned int* __restrict__ maxbits)
{
  __shared__ unsigned int h[2048];
  for (int i = threadIdx.x; i < 2048; i += blockDim.x) h[i] = 0u;
  __syncthreads();
  float mv = __uint_as_float(*maxbits);
  float scale = mv > 0.f ? 2048.f / mv : 0.f;
  for (int i = blockIdx.x * blockDim.x + threadIdx.x; i < n; i += gridDim.x * blockDim.x) {
    int bnum = (int)(f[i] * scale);
    if (bnum > 2047) bnum = 2047;
    if (bnum < 0) bnum = 0;
    atomicAdd(&h[bnum], 1u);
  }
  __syncthreads();
  for (int i = threadIdx.x; i < 2048; i += blockDim.x)
    if (h[i]) atomicAdd(&hist[i], h[i]);
}

__global__ void quant_k(const unsigned int* __restrict__ hist,
                        const unsigned int* __restrict__ maxbits,
                        float* stats, int n)
{
  if (threadIdx.x != 0 || blockIdx.x != 0) return;
  float mv   = __uint_as_float(*maxbits);
  float binw = mv / 2048.f;
  float pos25 = 0.25f * (float)(n - 1);
  float pos75 = 0.75f * (float)(n - 1);
  float cum = 0.f, q25 = 0.f, q75 = mv;
  bool d25 = false, d75 = false;
  for (int i = 0; i < 2048; ++i) {
    float cnt = (float)hist[i];
    if (!d25 && cum + cnt > pos25) { q25 = ((float)i + (pos25 - cum) / (cnt > 0.f ? cnt : 1.f)) * binw; d25 = true; }
    if (!d75 && cum + cnt > pos75) { q75 = ((float)i + (pos75 - cum) / (cnt > 0.f ? cnt : 1.f)) * binw; d75 = true; }
    cum += cnt;
  }
  stats[1] = q25; stats[2] = q75;
}

// per-(b,c) unbiased global std of x; one block each
__global__ void gstd_k(const float* __restrict__ x, float* stats)
{
  constexpr int HWN = 512 * 512;
  __shared__ float ss[256], ss2[256];
  const float* px = x + (size_t)blockIdx.x * HWN;
  float s = 0.f, s2 = 0.f;
  for (int i = threadIdx.x; i < HWN; i += 256) { float v = px[i]; s += v; s2 += v * v; }
  ss[threadIdx.x] = s; ss2[threadIdx.x] = s2; __syncthreads();
  for (int st = 128; st > 0; st >>= 1) {
    if (threadIdx.x < st) { ss[threadIdx.x] += ss[threadIdx.x + st]; ss2[threadIdx.x] += ss2[threadIdx.x + st]; }
    __syncthreads();
  }
  if (threadIdx.x == 0) {
    float nf = (float)HWN;
    float var = (ss2[0] - ss[0] * ss[0] / nf) / (nf - 1.f);
    stats[4 + blockIdx.x] = sqrtf(var > 0.f ? var : 0.f);
  }
}

__global__ void final_k(const float* __restrict__ fused, const float* __restrict__ stats,
                        const float* __restrict__ cwraw, float* __restrict__ out)
{
  constexpr int HWN = 512 * 512;
  int i = blockIdx.x * 256 + threadIdx.x;
  if (i >= 8 * HWN) return;
  int b = i / HWN;
  float q25 = stats[1], q75 = stats[2];
  float iqr = q75 - q25;
  float lo0 = q25 - 0.5f * iqr, hi0 = q75 + 0.5f * iqr;
  // softmax over 4 channel weights
  float cw[4];
  float mx = cwraw[0];
  for (int c = 1; c < 4; ++c) mx = fmaxf(mx, cwraw[c]);
  float se = 0.f;
  for (int c = 0; c < 4; ++c) { cw[c] = __expf(cwraw[c] - mx); se += cw[c]; }
  float v = fused[i];
  float acc = 0.f;
  for (int c = 0; c < 4; ++c) {
    float g  = stats[4 + b * 4 + c];
    float gf = fminf(fmaxf(g * 5.f, 0.5f), 2.f);
    // std(x * cw_c) over (H,W) == cw_c * std(x)  (constant scale)
    float cf = fminf(fmaxf((cw[c] / se) * g * 2.f, 0.8f), 1.2f);
    float lo = lo0 * gf * cf, hi = hi0 * gf * cf;
    float nn = (v - lo) / (hi - lo);
    nn = fminf(fmaxf(nn, 0.f), 1.f);
    acc += 1.f / (1.f + __expf(3.f - 6.f * nn));
  }
  out[i] = 0.25f * acc;
}

// ---------------------------------------------------------------------------
extern "C" void kernel_launch(void* const* d_in, const int* in_sizes, int n_in,
                              void* d_out, int out_size, void* d_ws, size_t ws_size,
                              hipStream_t stream)
{
  (void)in_sizes; (void)n_in; (void)out_size; (void)ws_size;
  const float* x  = (const float*)d_in[0];
  const float* w1 = (const float*)d_in[1];
  const float* b1 = (const float*)d_in[2];
  const float* w2 = (const float*)d_in[3];
  const float* b2 = (const float*)d_in[4];
  const float* w3 = (const float*)d_in[5];
  const float* b3 = (const float*)d_in[6];
  const float* fw = (const float*)d_in[7];
  const float* fb = (const float*)d_in[8];
  const float* cwraw = (const float*)d_in[9];
  float* out = (float*)d_out;

  const int B = 8, H = 512, W = 512;
  const int HW = H * W;
  char* ws = (char*)d_ws;

  // region A (peak 134 MB): conv1 output t1 (f16), later reused for f (f32)
  size_t o_t1 = 0;
  size_t t1_bytes = (size_t)B * 32 * HW * 2;            // 134,217,728
  // region B (67 MB): conv2 output t2 (f16), later reused for box/hist buffers
  size_t o_t2 = o_t1 + t1_bytes;

  _Float16* t1 = (_Float16*)(ws + o_t1);
  _Float16* t2 = (_Float16*)(ws + o_t2);
  float*    f  = (float*)(ws + o_t1);                   // overlays t1 (67 MB < 134 MB)

  size_t o_hm    = o_t2;                                //  8*HW*4 = 8.4 MB
  size_t o_hm2   = o_hm    + (size_t)8 * HW * 4;
  size_t o_chan  = o_hm2   + (size_t)8 * HW * 4;        //  HW*4   = 1 MB
  size_t o_fused = o_chan  + (size_t)HW * 4;            //  B*HW*4 = 8.4 MB
  size_t o_hist  = o_fused + (size_t)B * HW * 4;        //  8 KB
  size_t o_stats = o_hist  + 2048 * 4;                  //  40 floats
  size_t o_wp1   = (o_stats + 40 * 4 + 255) & ~(size_t)255;  // 32 x 64  f16
  size_t o_wp2   = (o_wp1 + 32 * 64 * 2 + 255) & ~(size_t)255;  // 16 x 288 f16
  size_t o_wp3   = (o_wp2 + 16 * 288 * 2 + 255) & ~(size_t)255; // 16 x 160 f16

  float* hm      = (float*)(ws + o_hm);
  float* hm2     = (float*)(ws + o_hm2);
  float* chansum = (float*)(ws + o_chan);
  float* fused   = (float*)(ws + o_fused);
  unsigned int* hist = (unsigned int*)(ws + o_hist);
  float* stats   = (float*)(ws + o_stats);
  _Float16* wp1  = (_Float16*)(ws + o_wp1);
  _Float16* wp2  = (_Float16*)(ws + o_wp2);
  _Float16* wp3  = (_Float16*)(ws + o_wp3);

  dim3 blk(256);
  dim3 cblk(128);

  // Weight pre-pack (zero-padded, wmma k-ordering)
  prep_w<<<(32 * 64 + 255) / 256, blk, 0, stream>>>(w1, wp1, 32, 4, 64, 2, 32);
  prep_w<<<(16 * 288 + 255) / 256, blk, 0, stream>>>(w2, wp2, 16, 32, 288, 5, 16);
  prep_w<<<(16 * 160 + 255) / 256, blk, 0, stream>>>(w3, wp3, 8, 16, 160, 4, 16);

  // Feature extractor (WMMA implicit GEMM); SEGS = 512/64 = 8
  conv3x3_wmma<4, 32, float,    true,  true,  false><<<B * 2 * H * 8, cblk, 0, stream>>>(x,  wp1, b1, t1, nullptr);
  conv3x3_wmma<32, 16, _Float16, true,  true,  false><<<B * 1 * H * 8, cblk, 0, stream>>>(t1, wp2, b2, t2, nullptr);
  conv3x3_wmma<16, 8,  _Float16, false, false, true ><<<B * 1 * H * 8, cblk, 0, stream>>>(t2, wp3, b3, nullptr, f);

  // Multi-scale local-std maps + fusion accumulate
  fill_f32<<<(B * HW + 255) / 256, blk, 0, stream>>>(fused, B * HW, 0.f);
  const int wins[3] = {11, 25, 49};
  for (int b = 0; b < B; ++b) {
    for (int wi = 0; wi < 3; ++wi) {
      fill_f32<<<(HW + 255) / 256, blk, 0, stream>>>(chansum, HW, 0.f);
      box_h<<<16, blk, 0, stream>>>(f + (size_t)b * 8 * HW, hm, hm2, wins[wi]);
      box_v<<<16, blk, 0, stream>>>(hm, hm2, chansum, wins[wi]);
      combine_map<<<HW / 256, blk, 0, stream>>>(fused + (size_t)b * HW, chansum, fw, wi);
    }
  }
  bias_relu<<<(B * HW + 255) / 256, blk, 0, stream>>>(fused, fb, B * HW);

  // Global quantiles + per-(b,c) std + mask
  fill_f32<<<1, 64, 0, stream>>>(stats, 40, 0.f);
  max_reduce<<<2048, blk, 0, stream>>>(fused, B * HW, (unsigned int*)stats);
  fill_f32<<<8, blk, 0, stream>>>((float*)hist, 2048, 0.f);
  hist_k<<<2048, blk, 0, stream>>>(fused, B * HW, hist, (unsigned int*)stats);
  quant_k<<<1, 32, 0, stream>>>(hist, (unsigned int*)stats, stats, B * HW);
  gstd_k<<<32, blk, 0, stream>>>(x, stats);
  final_k<<<(B * HW + 255) / 256, blk, 0, stream>>>(fused, stats, cwraw, out);
}
